// MultiHeadAttention_90907277787844
// MI455X (gfx1250) — matrix operand order repaired and checked by reading, not compiled, not run on hardware
//
#include <hip/hip_runtime.h>

// ---------------------------------------------------------------------------
// Flash-attention forward for gfx1250 (CDNA5, wave32, WMMA bf16 16x16x32).
//   B=2, H=16, S=2048, D=1024 (hd=64), fp32 in/out, bool mask -> -1e4 bias.
// One workgroup = 8 waves = 128 q-rows of one (b,h); KV streamed in 128-row
// tiles through LDS (bf16). 1/sqrt(hd) folded into the Q->bf16 conversion
// (exact power of two). Online softmax in fp32 with v_permlane16 butterfly
// reductions. P restaged via wave-private LDS (in-order DS, no block barrier)
// to convert WMMA C-layout -> A-layout.
// ---------------------------------------------------------------------------

typedef __attribute__((ext_vector_type(16))) __bf16 v16bf;
typedef __attribute__((ext_vector_type(8)))  __bf16 v8bf;
typedef __attribute__((ext_vector_type(4)))  __bf16 v4bf;
typedef __attribute__((ext_vector_type(8)))  float  v8f;

#define NHEAD   16
#define HD      64
#define SLEN    2048
#define DMODEL  1024
#define BATCH   2
#define QTILE   128
#define KTILE   128
#define NWAVES  8
#define NT      (SLEN / KTILE)
#define NSUB    (KTILE / 16)      // 8 score n-subtiles
#define NCHUNK  (KTILE / 32)      // 4 K-chunks for P*V

#define SHUF16(lo, hi) __builtin_shufflevector(lo, hi, 0,1,2,3,4,5,6,7,8,9,10,11,12,13,14,15)

// xor-butterfly within each 16-lane half via v_permlane16_b32 (pure VALU).
__device__ __forceinline__ float bfly16(float x, unsigned s1, unsigned s2, int xm) {
#if __has_builtin(__builtin_amdgcn_permlane16)
  int i = __builtin_bit_cast(int, x);
  i = __builtin_amdgcn_permlane16(i, i, (int)s1, (int)s2, false, false);
  return __builtin_bit_cast(float, i);
#else
  return __shfl_xor(x, xm, 32);
#endif
}
__device__ __forceinline__ float redmax16(float x) {
  x = fmaxf(x, bfly16(x, 0x67452301u, 0xEFCDAB89u, 1));
  x = fmaxf(x, bfly16(x, 0x54761032u, 0xDCFE98BAu, 2));
  x = fmaxf(x, bfly16(x, 0x32107654u, 0xBA98FEDCu, 4));
  x = fmaxf(x, bfly16(x, 0xFEDCBA98u, 0x76543210u, 8));
  return x;
}
__device__ __forceinline__ float redsum16(float x) {
  x += bfly16(x, 0x67452301u, 0xEFCDAB89u, 1);
  x += bfly16(x, 0x54761032u, 0xDCFE98BAu, 2);
  x += bfly16(x, 0x32107654u, 0xBA98FEDCu, 4);
  x += bfly16(x, 0xFEDCBA98u, 0x76543210u, 8);
  return x;
}
// byte -> float; pattern-matches to v_cvt_f32_ubyte{0..3}
__device__ __forceinline__ float ubyte_f(unsigned w, int b) {
  return (float)((w >> (8 * b)) & 0xFFu);
}

__global__ __launch_bounds__(256) void fa_fwd_bf16_wmma(
    const float* __restrict__ q, const float* __restrict__ k,
    const float* __restrict__ v, const unsigned char* __restrict__ mask,
    float* __restrict__ out) {

  const int tid  = threadIdx.x;
  const int lane = tid & 31;
  const int w    = tid >> 5;        // wave id 0..7
  const int g    = lane >> 4;       // lane half-group 0/1
  const int mrow = lane & 15;       // lane-within-half

  const int bid = blockIdx.x;
  const int qb  = bid & 15;                 // q-tile (S/128 = 16)
  const int h   = (bid >> 4) & (NHEAD - 1); // head
  const int b   = bid >> 8;                 // batch
  const int q0  = qb * QTILE;

  // LDS (row strides 16B-aligned; pads chosen for conflict-free b128 reads)
  __shared__ __bf16 sK   [KTILE][HD + 8];           // (k, d)        18.0 KB
  __shared__ __bf16 sVt  [HD][KTILE + 8];           // (d, k)        17.0 KB
  __shared__ float  sBias[QTILE][KTILE + 4];        // mask 0/1      66.0 KB
  __shared__ __bf16 sP   [NWAVES][16][KTILE + 8];   // P restage     34.0 KB

  // ---- Q A-fragments (16x64 = two 16x32 chunks), ISA A-layout; the
  //      1/sqrt(64) = 0.125 score scale is folded in here (exact in bf16).
  const size_t qoff = ((size_t)b * SLEN + (size_t)(q0 + w * 16 + mrow)) * DMODEL + h * HD;
  v16bf aq[2];
#pragma unroll
  for (int c = 0; c < 2; ++c)
#pragma unroll
    for (int h8 = 0; h8 < 2; ++h8) {
      const float* qp = q + qoff + c * 32 + g * 8 + h8 * 16;
#pragma unroll
      for (int t = 0; t < 8; ++t) aq[c][h8 * 8 + t] = (__bf16)(qp[t] * 0.125f);
    }

  // ---- softmax state (lane tracks its 8 C-layout rows) + O accumulators
  float rowm[8], rowl[8];
#pragma unroll
  for (int r = 0; r < 8; ++r) { rowm[r] = -3.0e38f; rowl[r] = 0.0f; }
  v8f acc[4];
#pragma unroll
  for (int n = 0; n < 4; ++n)
#pragma unroll
    for (int r = 0; r < 8; ++r) acc[n][r] = 0.0f;

  const int qlrow = w * 16 + g * 8;  // + r -> this lane's local q-row

  for (int kt = 0; kt < NT; ++kt) {
    const int kv0 = kt * KTILE;
    __syncthreads();  // previous iteration's LDS consumers done

    // ---- stage K tile: 128x64 fp32 -> bf16, packed b64 stores
#pragma unroll
    for (int i = 0; i < 8; ++i) {
      const int e = i * 256 + tid;            // 2048 float4's
      const int row = e >> 4, c4 = e & 15;
      const float4 kk = *(const float4*)(k +
          ((size_t)b * SLEN + (size_t)(kv0 + row)) * DMODEL + h * HD + c4 * 4);
      v4bf pk;
      pk[0] = (__bf16)kk.x; pk[1] = (__bf16)kk.y;
      pk[2] = (__bf16)kk.z; pk[3] = (__bf16)kk.w;
      *(v4bf*)&sK[row][c4 * 4] = pk;
    }
    // ---- stage V transposed: coalesced float2 loads (d-pairs x 4 k-rows),
    //      packed b64 stores into two sVt rows
#pragma unroll
    for (int i = 0; i < 4; ++i) {
      const int e = i * 256 + tid;            // 1024 tasks (32 d-pairs x 32 kg)
      const int dp = e & 31, kg = e >> 5;     // d = 2*dp, kg in 0..31
      const float* vp = v + ((size_t)b * SLEN + (size_t)(kv0 + kg * 4)) * DMODEL
                          + h * HD + 2 * dp;
      v4bf pk0, pk1;
#pragma unroll
      for (int t = 0; t < 4; ++t) {
        const float2 vv = *(const float2*)(vp + (size_t)t * DMODEL);
        pk0[t] = (__bf16)vv.x;
        pk1[t] = (__bf16)vv.y;
      }
      *(v4bf*)&sVt[2 * dp + 0][kg * 4] = pk0;
      *(v4bf*)&sVt[2 * dp + 1][kg * 4] = pk1;
    }
    // ---- stage mask -> float 0/1 (v_cvt_f32_ubyteN), b128 stores;
    //      the -1e4 factor is applied at score time via fma
#pragma unroll
    for (int i = 0; i < 4; ++i) {
      const int e = i * 256 + tid;            // 1024 x 16-byte chunks
      const int row = e >> 3, c16 = e & 7;
      const unsigned* mp = (const unsigned*)(mask +
          ((size_t)b * SLEN + (size_t)(q0 + row)) * SLEN + kv0 + c16 * 16);
#pragma unroll
      for (int wd = 0; wd < 4; ++wd) {
        const unsigned mw = mp[wd];
        float4 f;
        f.x = ubyte_f(mw, 0);
        f.y = ubyte_f(mw, 1);
        f.z = ubyte_f(mw, 2);
        f.w = ubyte_f(mw, 3);
        *(float4*)&sBias[row][c16 * 16 + wd * 4] = f;
      }
    }
    __syncthreads();

    // ---- prefetch next KV tile into L2 (uniform branch; global_prefetch_b8)
    if (kt + 1 < NT) {
      const int pr = tid & 127;
      const size_t poff = ((size_t)b * SLEN + (size_t)(kv0 + KTILE + pr)) * DMODEL + h * HD;
      __builtin_prefetch(k + poff, 0, 1);
      __builtin_prefetch(v + poff, 0, 1);
    }

    // ---- scores: S = (Q/8)*K^T, 8 n-subtiles x 2 K-chunks; mask via fma
    v8f sc[NSUB];
#pragma unroll
    for (int j = 0; j < NSUB; ++j) {
      const int nrow = j * 16 + mrow;  // B-layout: lane = col n, K-runs g*8..+7 and 16+g*8..+7
      const v8bf lo0 = *(const v8bf*)&sK[nrow][g * 8];
      const v8bf hi0 = *(const v8bf*)&sK[nrow][16 + g * 8];
      const v8bf lo1 = *(const v8bf*)&sK[nrow][32 + g * 8];
      const v8bf hi1 = *(const v8bf*)&sK[nrow][32 + 16 + g * 8];
      const v16bf b0 = SHUF16(lo0, hi0);
      const v16bf b1 = SHUF16(lo1, hi1);
      v8f t;
#pragma unroll
      for (int r = 0; r < 8; ++r) t[r] = 0.0f;
      t = __builtin_amdgcn_wmma_f32_16x16x32_bf16(false, aq[0], false, b0, (short)0, t, false, false);
      t = __builtin_amdgcn_wmma_f32_16x16x32_bf16(false, aq[1], false, b1, (short)0, t, false, false);
#pragma unroll
      for (int r = 0; r < 8; ++r)
        t[r] = __builtin_fmaf(sBias[qlrow + r][j * 16 + mrow], -10000.0f, t[r]);
      sc[j] = t;
    }

    // ---- online softmax (row stats via permlane16 butterflies)
    float corr[8];
#pragma unroll
    for (int r = 0; r < 8; ++r) {
      float tm = sc[0][r];
#pragma unroll
      for (int j = 1; j < NSUB; ++j) tm = fmaxf(tm, sc[j][r]);
      tm = redmax16(tm);
      const float nm = fmaxf(rowm[r], tm);
      corr[r] = exp2f((rowm[r] - nm) * 1.44269504f);
      rowm[r] = nm;
      float ls = 0.0f;
#pragma unroll
      for (int j = 0; j < NSUB; ++j) {
        const float pv = exp2f((sc[j][r] - nm) * 1.44269504f);
        sc[j][r] = pv;
        ls += pv;
      }
      ls = redsum16(ls);
      rowl[r] = rowl[r] * corr[r] + ls;
    }
#pragma unroll
    for (int n = 0; n < 4; ++n)
#pragma unroll
      for (int r = 0; r < 8; ++r) acc[n][r] *= corr[r];

    // ---- restage P (wave-private; DS in-order per wave -> no block barrier)
#pragma unroll
    for (int j = 0; j < NSUB; ++j)
#pragma unroll
      for (int r = 0; r < 8; ++r)
        sP[w][r + 8 * g][j * 16 + mrow] = (__bf16)sc[j][r];
    asm volatile("" ::: "memory");  // keep compiler from reordering DS ops

    v16bf pa[NCHUNK];
#pragma unroll
    for (int c = 0; c < NCHUNK; ++c) {
      const v8bf lo = *(const v8bf*)&sP[w][mrow][c * 32 + g * 8];
      const v8bf hi = *(const v8bf*)&sP[w][mrow][c * 32 + 16 + g * 8];
      pa[c] = SHUF16(lo, hi);
    }
    // ---- O += P*V  (4 out-dim subtiles x 4 K-chunks)
#pragma unroll
    for (int n = 0; n < 4; ++n)
#pragma unroll
      for (int c = 0; c < NCHUNK; ++c) {
        const v8bf lo = *(const v8bf*)&sVt[n * 16 + mrow][c * 32 + g * 8];
        const v8bf hi = *(const v8bf*)&sVt[n * 16 + mrow][c * 32 + 16 + g * 8];
        const v16bf bb = SHUF16(lo, hi);
        acc[n] = __builtin_amdgcn_wmma_f32_16x16x32_bf16(false, pa[c], false, bb,
                                                         (short)0, acc[n], false, false);
      }
  }

  // ---- normalize + store: out[b, q, h*64 + n*16 + mrow]
#pragma unroll
  for (int r = 0; r < 8; ++r) {
    const float inv = 1.0f / rowl[r];
    float* op = out + ((size_t)b * SLEN + (size_t)(q0 + w * 16 + g * 8 + r)) * DMODEL + h * HD;
#pragma unroll
    for (int n = 0; n < 4; ++n) op[n * 16 + mrow] = acc[n][r] * inv;
  }
}

extern "C" void kernel_launch(void* const* d_in, const int* in_sizes, int n_in,
                              void* d_out, int out_size, void* d_ws, size_t ws_size,
                              hipStream_t stream) {
  (void)in_sizes; (void)n_in; (void)d_ws; (void)ws_size; (void)out_size;
  const float*         q    = (const float*)d_in[0];
  const float*         k    = (const float*)d_in[1];
  const float*         v    = (const float*)d_in[2];
  const unsigned char* mask = (const unsigned char*)d_in[3];  // jax bool = 1 byte
  float* out = (float*)d_out;

  dim3 grid(BATCH * NHEAD * (SLEN / QTILE));  // 512 workgroups
  dim3 block(256);                            // 8 waves (wave32)
  hipLaunchKernelGGL(fa_fwd_bf16_wmma, grid, block, 0, stream, q, k, v, mask, out);
}
